// FourierFTLayer_26216480375132
// MI455X (gfx1250) — matrix operand order repaired and checked by reading, not compiled, not run on hardware
//
#include <hip/hip_runtime.h>

// ---------------------------------------------------------------------------
// FourierFT layer for MI455X (gfx1250, wave32, WMMA)
//   dW = real(ifft2(scatter(spectrum)))*150  ==  rank-2*nfreq product A*Bt^T
//   Weff = W + A*Bt^T         (bf16 WMMA, f32 accum, C preloaded with W)
//   y    = x * Weff^T + bias  (split-bf16 WMMA: hi*hi + hi*lo + lo*hi)
// LDS tiles stored [row][K], 16B-aligned rows -> fragments are ds_load_b128.
// Peeled double-buffered pipeline with scalar staging registers (no arrays /
// lambdas -> no scratch traffic): global loads of tile k+1 overlap WMMAs of k.
// ---------------------------------------------------------------------------

typedef __attribute__((ext_vector_type(16))) __bf16 v16bf;
typedef __attribute__((ext_vector_type(8)))  float  v8f;

#define IN_F   4096
#define OUT_F  4096
#define KPAD   2048          // 2*n_freq rounded up; zero padded
#define SCALING 150.0f

#define BM 128
#define BN 128
#define BK 32
#define PAD 8                // row stride 40 bf16 = 80B: 16B aligned, banks spread
#define LDS_STRIDE (BK + PAD)

__device__ __forceinline__ unsigned short f2bf_rne(float f) {
  unsigned int u = __float_as_uint(f);
  unsigned int r = u + 0x7FFFu + ((u >> 16) & 1u);   // round-to-nearest-even
  return (unsigned short)(r >> 16);
}
__device__ __forceinline__ float bf2f(unsigned short h) {
  return __uint_as_float(((unsigned int)h) << 16);
}

union BF16Frag { v16bf v; uint4 q[2]; };

// Fragment gathers from an LDS tile stored [row][K] (pitch LDS_STRIDE).
// A (16x32 bf16, wave32): lane-half hl holds K in [8hl,8hl+8) then
// [16+8hl,...+8), element order = K ascending -> two aligned 16B chunks.
// B (32x16): same K order as A -> two aligned 16B chunks.
__device__ __forceinline__ v16bf frag_A(const unsigned short* rowp, int hl) {
  BF16Frag f;
  f.q[0] = *(const uint4*)(rowp + 8 * hl);
  f.q[1] = *(const uint4*)(rowp + 16 + 8 * hl);
  return f.v;
}
__device__ __forceinline__ v16bf frag_B(const unsigned short* rowp, int hl) {
  BF16Frag f;
  f.q[0] = *(const uint4*)(rowp + 8 * hl);
  f.q[1] = *(const uint4*)(rowp + 16 + 8 * hl);
  return f.v;
}

__device__ __forceinline__ ushort4 split_hi(float4 f) {
  ushort4 h;
  h.x = f2bf_rne(f.x); h.y = f2bf_rne(f.y);
  h.z = f2bf_rne(f.z); h.w = f2bf_rne(f.w);
  return h;
}
__device__ __forceinline__ ushort4 split_lo(float4 f, ushort4 h) {
  ushort4 l;
  l.x = f2bf_rne(f.x - bf2f(h.x)); l.y = f2bf_rne(f.y - bf2f(h.y));
  l.z = f2bf_rne(f.z - bf2f(h.z)); l.w = f2bf_rne(f.w - bf2f(h.w));
  return l;
}

// 8 WMMAs: one bf16 K=32 step for a 64x32 wave tile (4x2 16x16 tiles).
__device__ __forceinline__ void wmma_step_1x(
    v8f (&acc)[4][2],
    const unsigned short (*sAbuf)[LDS_STRIDE],
    const unsigned short (*sBbuf)[LDS_STRIDE],
    int waveM, int waveN, int l16, int hl) {
  v16bf b0 = frag_B(&sBbuf[waveN * 32 + l16][0], hl);
  v16bf b1 = frag_B(&sBbuf[waveN * 32 + 16 + l16][0], hl);
#pragma unroll
  for (int mt = 0; mt < 4; ++mt) {
    v16bf a = frag_A(&sAbuf[waveM * 64 + mt * 16 + l16][0], hl);
    acc[mt][0] = __builtin_amdgcn_wmma_f32_16x16x32_bf16(
        false, a, false, b0, (short)0, acc[mt][0], false, false);
    acc[mt][1] = __builtin_amdgcn_wmma_f32_16x16x32_bf16(
        false, a, false, b1, (short)0, acc[mt][1], false, false);
  }
}

// 24 WMMAs: one split-bf16 K=32 step (hi*hi + hi*lo + lo*hi).
__device__ __forceinline__ void wmma_step_3x(
    v8f (&acc)[4][2],
    const unsigned short (*Xh)[LDS_STRIDE], const unsigned short (*Xl)[LDS_STRIDE],
    const unsigned short (*Wh)[LDS_STRIDE], const unsigned short (*Wl)[LDS_STRIDE],
    int waveM, int waveN, int l16, int hl) {
  v16bf bh0 = frag_B(&Wh[waveN * 32 + l16][0], hl);
  v16bf bh1 = frag_B(&Wh[waveN * 32 + 16 + l16][0], hl);
  v16bf bl0 = frag_B(&Wl[waveN * 32 + l16][0], hl);
  v16bf bl1 = frag_B(&Wl[waveN * 32 + 16 + l16][0], hl);
#pragma unroll
  for (int mt = 0; mt < 4; ++mt) {
    int mrow = waveM * 64 + mt * 16 + l16;
    v16bf ah = frag_A(&Xh[mrow][0], hl);
    v16bf al = frag_A(&Xl[mrow][0], hl);
    acc[mt][0] = __builtin_amdgcn_wmma_f32_16x16x32_bf16(
        false, ah, false, bh0, (short)0, acc[mt][0], false, false);
    acc[mt][0] = __builtin_amdgcn_wmma_f32_16x16x32_bf16(
        false, ah, false, bl0, (short)0, acc[mt][0], false, false);
    acc[mt][0] = __builtin_amdgcn_wmma_f32_16x16x32_bf16(
        false, al, false, bh0, (short)0, acc[mt][0], false, false);
    acc[mt][1] = __builtin_amdgcn_wmma_f32_16x16x32_bf16(
        false, ah, false, bh1, (short)0, acc[mt][1], false, false);
    acc[mt][1] = __builtin_amdgcn_wmma_f32_16x16x32_bf16(
        false, ah, false, bl1, (short)0, acc[mt][1], false, false);
    acc[mt][1] = __builtin_amdgcn_wmma_f32_16x16x32_bf16(
        false, al, false, bh1, (short)0, acc[mt][1], false, false);
  }
}

// ---------------------------------------------------------------------------
// Kernel 1: A[o][kk] : kk=2k -> s_k*C*cos(2pi*r_k*o/4096), kk=2k+1 -> -s_k*C*sin
// ---------------------------------------------------------------------------
__global__ void gen_fourier_A(const float* __restrict__ spectrum,
                              const int* __restrict__ rowsIdx,
                              int nfreq, unsigned short* __restrict__ Agrid) {
  int idx = blockIdx.x * blockDim.x + threadIdx.x;      // OUT_F*KPAD total
  int o  = idx >> 11;                                   // / KPAD
  int kk = idx & (KPAD - 1);
  int k  = kk >> 1;
  float val = 0.0f;
  if (k < nfreq) {
    int r = rowsIdx[k];
    int phase = (r * o) & (IN_F - 1);                   // exact mod 4096
    float th = (float)phase * (6.28318530717958647692f / 4096.0f);
    float s, c;
    __sincosf(th, &s, &c);
    float amp = spectrum[k] * (SCALING / (4096.0f * 4096.0f));
    val = (kk & 1) ? (-amp * s) : (amp * c);
  }
  Agrid[idx] = f2bf_rne(val);
}

// ---------------------------------------------------------------------------
// Kernel 2 (transposed): Bt[i][kk] : kk=2k -> cos(2pi*c_k*i/4096), kk+1 -> sin
// ---------------------------------------------------------------------------
__global__ void gen_fourier_Bt(const int* __restrict__ colsIdx, int nfreq,
                               unsigned short* __restrict__ Bt) {
  int idx = blockIdx.x * blockDim.x + threadIdx.x;      // IN_F*KPAD total
  int i  = idx >> 11;                                   // / KPAD
  int kk = idx & (KPAD - 1);
  int k  = kk >> 1;
  float val = 0.0f;
  if (k < nfreq) {
    int cc = colsIdx[k];
    int phase = (cc * i) & (IN_F - 1);
    float th = (float)phase * (6.28318530717958647692f / 4096.0f);
    float s, c;
    __sincosf(th, &s, &c);
    val = (kk & 1) ? s : c;
  }
  Bt[idx] = f2bf_rne(val);
}

// ---------------------------------------------------------------------------
// Kernel 3: Weff[m][n] = W[m][n] + sum_k A[m][k]*Bt[n][k]   (bf16 WMMA)
// ---------------------------------------------------------------------------
#define W_LOAD(K0)                                                            \
  ra0 = *(const uint4*)(gA0 + (K0));                                          \
  ra1 = *(const uint4*)(gA0 + 64 * KPAD + (K0));                              \
  rb0 = *(const uint4*)(gB0 + (K0));                                          \
  rb1 = *(const uint4*)(gB0 + 64 * KPAD + (K0));                              \
  if ((K0) + 2 * BK < KPAD) {                                                 \
    __builtin_prefetch(gA0 + (K0) + 2 * BK);                                  \
    __builtin_prefetch(gA0 + 64 * KPAD + (K0) + 2 * BK);                      \
    __builtin_prefetch(gB0 + (K0) + 2 * BK);                                  \
    __builtin_prefetch(gB0 + 64 * KPAD + (K0) + 2 * BK);                      \
  }

#define W_STORE(BUF)                                                          \
  *(uint4*)&sA[BUF][rr][cc]      = ra0;                                       \
  *(uint4*)&sA[BUF][rr + 64][cc] = ra1;                                       \
  *(uint4*)&sB[BUF][rr][cc]      = rb0;                                       \
  *(uint4*)&sB[BUF][rr + 64][cc] = rb1;

__global__ __launch_bounds__(256, 1)
void weff_gemm(const unsigned short* __restrict__ A,    // [OUT_F][KPAD]  bf16
               const unsigned short* __restrict__ Bt,   // [IN_F][KPAD]   bf16
               const float* __restrict__ W,             // [OUT_F][IN_F]
               float* __restrict__ Weff) {              // [OUT_F][IN_F]
  __shared__ unsigned short sA[2][BM][LDS_STRIDE];
  __shared__ unsigned short sB[2][BN][LDS_STRIDE];

  const int tid   = threadIdx.x;
  const int lane  = tid & 31;
  const int wave  = tid >> 5;
  const int waveM = wave >> 2;        // 0..1
  const int waveN = wave & 3;         // 0..3
  const int hl    = lane >> 4;        // lane half (wave32)
  const int l16   = lane & 15;

  const int m0 = blockIdx.y * BM;
  const int n0 = blockIdx.x * BN;

  // Staging coords: chunk j covers row rr + j*64, 8 bf16 at col cc.
  const int rr = tid >> 2, cc = (tid & 3) * 8;
  const unsigned short* gA0 = &A[(m0 + rr) * KPAD + cc];
  const unsigned short* gB0 = &Bt[(n0 + rr) * KPAD + cc];

  // Preload C accumulators with W per the 16x16 f32 C/D VGPR layout:
  //   lane half hl, VGPR j -> M = j + 8*hl ; N = l16
  v8f acc[4][2];
#pragma unroll
  for (int mt = 0; mt < 4; ++mt) {
    int mbase = m0 + waveM * 64 + mt * 16 + hl * 8;
#pragma unroll
    for (int nt = 0; nt < 2; ++nt) {
      int col = n0 + waveN * 32 + nt * 16 + l16;
      v8f a;
#pragma unroll
      for (int j = 0; j < 8; ++j) a[j] = W[(mbase + j) * IN_F + col];
      acc[mt][nt] = a;
    }
  }

  uint4 ra0, ra1, rb0, rb1;           // in-flight tile (scalars: no scratch)
  W_LOAD(0)
  W_STORE(0)
  __syncthreads();

  const int nIter = KPAD / BK;        // 64
  for (int it = 0; it < nIter - 1; ++it) {
    const int cur = it & 1;
    W_LOAD((it + 1) * BK)                       // overlap with WMMAs below
    wmma_step_1x(acc, sA[cur], sB[cur], waveM, waveN, l16, hl);
    W_STORE(cur ^ 1)
    __syncthreads();
  }
  wmma_step_1x(acc, sA[(nIter - 1) & 1], sB[(nIter - 1) & 1],
               waveM, waveN, l16, hl);

#pragma unroll
  for (int mt = 0; mt < 4; ++mt) {
    int mbase = m0 + waveM * 64 + mt * 16 + hl * 8;
#pragma unroll
    for (int nt = 0; nt < 2; ++nt) {
      int col = n0 + waveN * 32 + nt * 16 + l16;
#pragma unroll
      for (int j = 0; j < 8; ++j) Weff[(mbase + j) * IN_F + col] = acc[mt][nt][j];
    }
  }
}

// ---------------------------------------------------------------------------
// Kernel 4: Y[m][n] = sum_k X[m][k]*Weff[n][k] + bias[n]   (split-bf16 WMMA)
// ---------------------------------------------------------------------------
#define X_LOAD(K0)                                                            \
  rx0 = *(const float4*)(gx0 + (K0));                                         \
  rx1 = *(const float4*)(gx0 + 32 * IN_F + (K0));                             \
  rx2 = *(const float4*)(gx0 + 64 * IN_F + (K0));                             \
  rx3 = *(const float4*)(gx0 + 96 * IN_F + (K0));                             \
  rw0 = *(const float4*)(gw0 + (K0));                                         \
  rw1 = *(const float4*)(gw0 + 32 * IN_F + (K0));                             \
  rw2 = *(const float4*)(gw0 + 64 * IN_F + (K0));                             \
  rw3 = *(const float4*)(gw0 + 96 * IN_F + (K0));                             \
  if ((K0) + 2 * BK < IN_F) {                                                 \
    __builtin_prefetch(gx0 + (K0) + 2 * BK);                                  \
    __builtin_prefetch(gx0 + 64 * IN_F + (K0) + 2 * BK);                      \
    __builtin_prefetch(gw0 + (K0) + 2 * BK);                                  \
    __builtin_prefetch(gw0 + 64 * IN_F + (K0) + 2 * BK);                      \
  }

#define X_SPLIT1(V, DH, DL, ROW)                                              \
  {                                                                           \
    ushort4 h = split_hi(V);                                                  \
    *(ushort4*)&DH[ROW][cc] = h;                                              \
    *(ushort4*)&DL[ROW][cc] = split_lo(V, h);                                 \
  }

#define X_STORE(BUF)                                                          \
  X_SPLIT1(rx0, sXh[BUF], sXl[BUF], rr)                                       \
  X_SPLIT1(rx1, sXh[BUF], sXl[BUF], rr + 32)                                  \
  X_SPLIT1(rx2, sXh[BUF], sXl[BUF], rr + 64)                                  \
  X_SPLIT1(rx3, sXh[BUF], sXl[BUF], rr + 96)                                  \
  X_SPLIT1(rw0, sWh[BUF], sWl[BUF], rr)                                       \
  X_SPLIT1(rw1, sWh[BUF], sWl[BUF], rr + 32)                                  \
  X_SPLIT1(rw2, sWh[BUF], sWl[BUF], rr + 64)                                  \
  X_SPLIT1(rw3, sWh[BUF], sWl[BUF], rr + 96)

__global__ __launch_bounds__(256, 1)
void fused_linear(const float* __restrict__ X,          // [Mtot][IN_F]
                  const float* __restrict__ Weff,       // [OUT_F][IN_F]
                  const float* __restrict__ bias,       // [OUT_F]
                  float* __restrict__ Y, int Mtot) {
  __shared__ unsigned short sXh[2][BM][LDS_STRIDE];
  __shared__ unsigned short sXl[2][BM][LDS_STRIDE];
  __shared__ unsigned short sWh[2][BN][LDS_STRIDE];
  __shared__ unsigned short sWl[2][BN][LDS_STRIDE];

  const int tid   = threadIdx.x;
  const int lane  = tid & 31;
  const int wave  = tid >> 5;
  const int waveM = wave >> 2;
  const int waveN = wave & 3;
  const int hl    = lane >> 4;
  const int l16   = lane & 15;

  const int m0 = blockIdx.y * BM;
  const int n0 = blockIdx.x * BN;

  // Staging coords: chunk j covers row rr + j*32, 4 floats at col cc.
  const int rr = tid >> 3, cc = (tid & 7) * 4;
  const float* gx0 = &X[(m0 + rr) * IN_F + cc];
  const float* gw0 = &Weff[(n0 + rr) * IN_F + cc];

  v8f acc[4][2];
#pragma unroll
  for (int mt = 0; mt < 4; ++mt)
#pragma unroll
    for (int nt = 0; nt < 2; ++nt)
      acc[mt][nt] = (v8f){0.f,0.f,0.f,0.f,0.f,0.f,0.f,0.f};

  float4 rx0, rx1, rx2, rx3, rw0, rw1, rw2, rw3;   // in-flight tile (scalars)
  X_LOAD(0)
  X_STORE(0)
  __syncthreads();

  const int nIter = IN_F / BK;        // 128
  for (int it = 0; it < nIter - 1; ++it) {
    const int cur = it & 1;
    X_LOAD((it + 1) * BK)                       // overlap with WMMAs below
    wmma_step_3x(acc, sXh[cur], sXl[cur], sWh[cur], sWl[cur],
                 waveM, waveN, l16, hl);
    X_STORE(cur ^ 1)
    __syncthreads();
  }
  {
    const int last = (nIter - 1) & 1;
    wmma_step_3x(acc, sXh[last], sXl[last], sWh[last], sWl[last],
                 waveM, waveN, l16, hl);
  }

#pragma unroll
  for (int mt = 0; mt < 4; ++mt) {
    int mbase = m0 + waveM * 64 + mt * 16 + hl * 8;
#pragma unroll
    for (int nt = 0; nt < 2; ++nt) {
      int col = n0 + waveN * 32 + nt * 16 + l16;
      float b = bias[col];
#pragma unroll
      for (int j = 0; j < 8; ++j)
        Y[(mbase + j) * OUT_F + col] = acc[mt][nt][j] + b;
    }
  }
}

// ---------------------------------------------------------------------------
extern "C" void kernel_launch(void* const* d_in, const int* in_sizes, int n_in,
                              void* d_out, int out_size, void* d_ws, size_t ws_size,
                              hipStream_t stream) {
  const float* x        = (const float*)d_in[0];   // [4,2048,4096]
  const float* spectrum = (const float*)d_in[1];   // [nfreq]
  const int*   indices  = (const int*)d_in[2];     // [2,nfreq]
  const float* weight   = (const float*)d_in[3];   // [4096,4096]
  const float* bias     = (const float*)d_in[4];   // [4096]
  float* out = (float*)d_out;

  const int nfreq = in_sizes[1];
  const int Mtot  = in_sizes[0] / IN_F;            // 8192

  // workspace layout: A bf16 (16MB) | Bt bf16 (16MB) | Weff f32 (64MB)
  unsigned short* Agrid  = (unsigned short*)d_ws;
  unsigned short* Btgrid = Agrid + (size_t)OUT_F * KPAD;
  float*          Weff   = (float*)(Btgrid + (size_t)IN_F * KPAD);

  const int* rowsIdx = indices;
  const int* colsIdx = indices + nfreq;

  gen_fourier_A<<<(OUT_F * KPAD) / 256, 256, 0, stream>>>(spectrum, rowsIdx, nfreq, Agrid);
  gen_fourier_Bt<<<(IN_F * KPAD) / 256, 256, 0, stream>>>(colsIdx, nfreq, Btgrid);
  weff_gemm<<<dim3(IN_F / BN, OUT_F / BM), 256, 0, stream>>>(Agrid, Btgrid, weight, Weff);
  fused_linear<<<dim3(OUT_F / BN, Mtot / BM), 256, 0, stream>>>(x, Weff, bias, out, Mtot);
}
